// LocalSelfAttentionBlock_9912784519507
// MI455X (gfx1250) — compile-verified, matrix operand order checked by reading, not run on hardware
//
#include <hip/hip_runtime.h>
#include <hip/hip_bf16.h>

typedef __attribute__((ext_vector_type(16))) _Float16 v16h;
typedef __attribute__((ext_vector_type(8)))  _Float16 v8h;
typedef __attribute__((ext_vector_type(8)))  float    v8f;

#define DEV __device__ __forceinline__

// ---------------------------------------------------------------------------
// Async global->LDS copy path (CDNA5 GLOBAL_LOAD_ASYNC_TO_LDS_B128, ASYNCcnt)
// Guarded: falls back to direct global fragment loads if builtins are absent.
// Probe round 2 showed the builtin exists with signature
//   (v4i __as1* src, v4i __as3* dst, imm int offset, imm int cpol)
// ---------------------------------------------------------------------------
#if defined(__has_builtin)
#if __has_builtin(__builtin_amdgcn_global_load_async_to_lds_b128)
#define HAVE_ASYNC_LDS 1
#endif
#endif
#ifndef HAVE_ASYNC_LDS
#define HAVE_ASYNC_LDS 0
#endif

#if HAVE_ASYNC_LDS
typedef int v4i_vs __attribute__((vector_size(16)));
typedef __attribute__((address_space(1))) v4i_vs* gptr128_t;
typedef __attribute__((address_space(3))) v4i_vs* lptr128_t;

DEV void async_copy_b128(const _Float16* gsrc, _Float16* ldst) {
  _Float16* g = (_Float16*)gsrc;  // drop const first
  __builtin_amdgcn_global_load_async_to_lds_b128((gptr128_t)g, (lptr128_t)ldst,
                                                 0, 0);
}
DEV void async_wait0() {
#if __has_builtin(__builtin_amdgcn_s_wait_asynccnt)
  __builtin_amdgcn_s_wait_asynccnt(0);
#else
  asm volatile("s_wait_asynccnt 0" ::: "memory");
#endif
}
#endif

// ---------------------------------------------------------------------------
// Fragment helpers (CDNA5 WMMA 16x16x32 f16, wave32).
// A-matrix 16x32 (MxK): lanes 0-15 row=lane, halves 0..7 => K = 0..7,
// halves 8..15 => K = 16..23; lanes 16-31 same rows, K offset +8.
// The same loader serves B (32x16, KxN) when the source is stored [N, K]
// row-major (lane = n, halves = K chunks) — true for W^T GEMMs and for K^T.
// ---------------------------------------------------------------------------
DEV v16h load_frag16x32(const _Float16* __restrict__ base, int stride) {
  unsigned lane = threadIdx.x & 31u;
  unsigned row  = lane & 15u;
  unsigned ko   = (lane >> 4) * 8u;
  const _Float16* p = base + (size_t)row * (size_t)stride + ko;
  v8h lo = *(const v8h*)(p);
  v8h hi = *(const v8h*)(p + 16);
  v16h f;
#pragma unroll
  for (int i = 0; i < 8; ++i) { f[i] = lo[i]; f[i + 8] = hi[i]; }
  return f;
}

DEV v8f wmma_f16(v16h a, v16h b, v8f c) {
  return __builtin_amdgcn_wmma_f32_16x16x32_f16(false, a, false, b,
                                                (short)0, c, false, false);
}

// ---------------------------------------------------------------------------
// fp32 -> fp16 convert
// ---------------------------------------------------------------------------
__global__ __launch_bounds__(256) void cvt_kernel(const float* __restrict__ src,
                                                  _Float16* __restrict__ dst,
                                                  int n) {
  int i = blockIdx.x * 256 + threadIdx.x;
  if (i < n) dst[i] = (_Float16)src[i];
}

// ---------------------------------------------------------------------------
// QKV GEMM: Out[16384,3072] = X[16384,1024] * Wt[3072,1024]^T + bias
// block tile 128x128, 8 waves, wave tile 32x64 (2x4 WMMA accumulators)
// ---------------------------------------------------------------------------
__global__ __launch_bounds__(256) void qkv_gemm_kernel(
    const _Float16* __restrict__ X, const _Float16* __restrict__ Wt,
    const float* __restrict__ bias, _Float16* __restrict__ Out) {
  const int K = 1024, N = 3072;
  int wave = threadIdx.x >> 5;
  int m0 = blockIdx.x * 128 + (wave & 3) * 32;
  int n0 = blockIdx.y * 128 + (wave >> 2) * 64;

  v8f zero8 = {0.f, 0.f, 0.f, 0.f, 0.f, 0.f, 0.f, 0.f};
  v8f acc[2][4];
#pragma unroll
  for (int i = 0; i < 2; ++i)
#pragma unroll
    for (int j = 0; j < 4; ++j) acc[i][j] = zero8;

  const _Float16* x0 = X + (size_t)m0 * K;
  const _Float16* x1 = X + (size_t)(m0 + 16) * K;
  for (int k0 = 0; k0 < K; k0 += 32) {
    __builtin_prefetch(x0 + k0 + 128, 0, 1);
    v16h a0 = load_frag16x32(x0 + k0, K);
    v16h a1 = load_frag16x32(x1 + k0, K);
#pragma unroll
    for (int j = 0; j < 4; ++j) {
      v16h b = load_frag16x32(Wt + (size_t)(n0 + 16 * j) * K + k0, K);
      acc[0][j] = wmma_f16(a0, b, acc[0][j]);
      acc[1][j] = wmma_f16(a1, b, acc[1][j]);
    }
  }

  unsigned lane = threadIdx.x & 31u;
  int cn = lane & 15;
  int rm = (lane >> 4) * 8;
#pragma unroll
  for (int i = 0; i < 2; ++i)
#pragma unroll
    for (int j = 0; j < 4; ++j) {
      int mm = m0 + i * 16 + rm;
      int nn = n0 + j * 16 + cn;
      float bz = bias[nn];
#pragma unroll
      for (int r = 0; r < 8; ++r)
        Out[(size_t)(mm + r) * N + nn] = (_Float16)(acc[i][j][r] + bz);
    }
}

// ---------------------------------------------------------------------------
// Windowed flash attention. One block = (b, window, head, 128-query tile).
// Wave handles 16 query rows; streams 32 keys per step:
//   - K chunk staged once per block into LDS via async-to-LDS (ASYNCcnt)
//   - 8 WMMAs QK^T, online softmax (wave32 shfl_xor row reductions)
//   - P repacked via per-wave LDS tile, V staged transposed in LDS
//   - 8 WMMAs PV
// ---------------------------------------------------------------------------
__global__ __launch_bounds__(256) void attn_kernel(
    const _Float16* __restrict__ QKV, const unsigned char* __restrict__ kpm,
    _Float16* __restrict__ Oh) {
  const int TD = 3072;
  int id = blockIdx.x;
  int qt = id & 7, h = (id >> 3) & 7, win = (id >> 6) & 7, b = id >> 9;
  int wave = threadIdx.x >> 5;
  unsigned lane = threadIdx.x & 31u;
  int cn = lane & 15;
  int rm = (lane >> 4) * 8;
  int rowbase = b * 8192 + win * 1024;
  int q0 = qt * 128 + wave * 16;

  const _Float16* Qp = QKV + (size_t)(rowbase + q0) * TD + h * 128;
  const _Float16* Kp = QKV + (size_t)rowbase * TD + 1024 + h * 128;
  const _Float16* Vp = QKV + (size_t)rowbase * TD + 2048 + h * 128;

  __shared__ __align__(16) _Float16 ldsP[8 * 16 * 40];   // per-wave P 16x32
  __shared__ __align__(16) _Float16 ldsV[128 * 40];      // V^T chunk [c][k]
#if HAVE_ASYNC_LDS
  __shared__ __align__(16) _Float16 ldsK[32 * 136];      // K chunk [k][c] pad 8
#endif

  v16h qf[4];
#pragma unroll
  for (int kk = 0; kk < 4; ++kk) qf[kk] = load_frag16x32(Qp + kk * 32, TD);

  v8f zero8 = {0.f, 0.f, 0.f, 0.f, 0.f, 0.f, 0.f, 0.f};
  v8f o[8];
#pragma unroll
  for (int t = 0; t < 8; ++t) o[t] = zero8;
  float mrun[8], lrun[8];
#pragma unroll
  for (int r = 0; r < 8; ++r) { mrun[r] = -3.0e38f; lrun[r] = 0.f; }

  const float scale = 0.08838834764831845f;  // 1/sqrt(128)

  for (int kc = 0; kc < 1024; kc += 32) {
    __syncthreads();

#if HAVE_ASYNC_LDS
    // stage K chunk (32 keys x 128 ch) via async DMA: 512 b128 chunks,
    // 2 per thread; no VGPR round-trip, tracked by ASYNCcnt.
#pragma unroll
    for (int i = 0; i < 2; ++i) {
      int ch = threadIdx.x + i * 256;     // 0..511
      int key = ch >> 4;                  // 16 chunks per key row
      int c8 = (ch & 15) * 8;             // half offset within row
      async_copy_b128(Kp + (size_t)(kc + key) * TD + c8, ldsK + key * 136 + c8);
    }
#endif

    {  // stage V chunk transposed: [channel][key], key-dim contiguous
      int k = threadIdx.x >> 3;
      int c0 = (threadIdx.x & 7) * 16;
      const v8h* src = (const v8h*)(Vp + (size_t)(kc + k) * TD + c0);
      v8h v0 = src[0], v1 = src[1];
#pragma unroll
      for (int i = 0; i < 8; ++i) {
        ldsV[(c0 + i) * 40 + k]     = v0[i];
        ldsV[(c0 + 8 + i) * 40 + k] = v1[i];
      }
    }

#if HAVE_ASYNC_LDS
    async_wait0();
#endif
    __syncthreads();

    v8f s[2];
    s[0] = zero8; s[1] = zero8;
#pragma unroll
    for (int sub = 0; sub < 2; ++sub)
#pragma unroll
      for (int kk = 0; kk < 4; ++kk) {
#if HAVE_ASYNC_LDS
        v16h bk = load_frag16x32(ldsK + (sub * 16) * 136 + kk * 32, 136);
#else
        v16h bk = load_frag16x32(Kp + (size_t)(kc + sub * 16) * TD + kk * 32, TD);
#endif
        s[sub] = wmma_f16(qf[kk], bk, s[sub]);
      }

    // scale + key-padding mask
#pragma unroll
    for (int sub = 0; sub < 2; ++sub) {
      bool pm = kpm[rowbase + kc + sub * 16 + cn] != 0;
#pragma unroll
      for (int r = 0; r < 8; ++r) {
        float x = s[sub][r] * scale;
        s[sub][r] = pm ? -1.0e9f : x;
      }
    }

    // row max across the 16-lane half (rows live per half-wave)
    float tm[8];
#pragma unroll
    for (int r = 0; r < 8; ++r) tm[r] = fmaxf(s[0][r], s[1][r]);
#pragma unroll
    for (int off = 1; off < 16; off <<= 1)
#pragma unroll
      for (int r = 0; r < 8; ++r) tm[r] = fmaxf(tm[r], __shfl_xor(tm[r], off, 32));

    float corr[8], rs[8];
#pragma unroll
    for (int r = 0; r < 8; ++r) {
      float nm = fmaxf(mrun[r], tm[r]);
      corr[r] = __expf(mrun[r] - nm);
      mrun[r] = nm;
      rs[r] = 0.f;
    }
#pragma unroll
    for (int sub = 0; sub < 2; ++sub)
#pragma unroll
      for (int r = 0; r < 8; ++r) {
        float pv = __expf(s[sub][r] - mrun[r]);
        s[sub][r] = pv;
        rs[r] += pv;
      }
#pragma unroll
    for (int off = 1; off < 16; off <<= 1)
#pragma unroll
      for (int r = 0; r < 8; ++r) rs[r] += __shfl_xor(rs[r], off, 32);
#pragma unroll
    for (int r = 0; r < 8; ++r) lrun[r] = lrun[r] * corr[r] + rs[r];
#pragma unroll
    for (int t = 0; t < 8; ++t)
#pragma unroll
      for (int r = 0; r < 8; ++r) o[t][r] *= corr[r];

    // repack P (C layout) -> A-fragment layout via per-wave LDS tile
    _Float16* pb = ldsP + wave * (16 * 40);
#pragma unroll
    for (int sub = 0; sub < 2; ++sub)
#pragma unroll
      for (int r = 0; r < 8; ++r)
        pb[(rm + r) * 40 + sub * 16 + cn] = (_Float16)s[sub][r];

    v16h pf = load_frag16x32(pb, 40);
#pragma unroll
    for (int t = 0; t < 8; ++t) {
      v16h bv = load_frag16x32(ldsV + t * 16 * 40, 40);
      o[t] = wmma_f16(pf, bv, o[t]);
    }
  }

#pragma unroll
  for (int r = 0; r < 8; ++r) {
    float invl = 1.0f / lrun[r];
    _Float16* dst = Oh + (size_t)(rowbase + q0 + rm + r) * 1024 + h * 128 + cn;
#pragma unroll
    for (int t = 0; t < 8; ++t) dst[t * 16] = (_Float16)(o[t][r] * invl);
  }
}

// ---------------------------------------------------------------------------
// Output projection + bias + residual (f32 out)
// ---------------------------------------------------------------------------
__global__ __launch_bounds__(256) void proj_gemm_kernel(
    const _Float16* __restrict__ A, const _Float16* __restrict__ Wt,
    const float* __restrict__ bias, const float* __restrict__ resid,
    float* __restrict__ Out) {
  const int K = 1024, N = 1024;
  int wave = threadIdx.x >> 5;
  int m0 = blockIdx.x * 128 + (wave & 3) * 32;
  int n0 = blockIdx.y * 128 + (wave >> 2) * 64;

  v8f zero8 = {0.f, 0.f, 0.f, 0.f, 0.f, 0.f, 0.f, 0.f};
  v8f acc[2][4];
#pragma unroll
  for (int i = 0; i < 2; ++i)
#pragma unroll
    for (int j = 0; j < 4; ++j) acc[i][j] = zero8;

  const _Float16* a0 = A + (size_t)m0 * K;
  const _Float16* a1 = A + (size_t)(m0 + 16) * K;
  for (int k0 = 0; k0 < K; k0 += 32) {
    __builtin_prefetch(a0 + k0 + 128, 0, 1);
    v16h f0 = load_frag16x32(a0 + k0, K);
    v16h f1 = load_frag16x32(a1 + k0, K);
#pragma unroll
    for (int j = 0; j < 4; ++j) {
      v16h b = load_frag16x32(Wt + (size_t)(n0 + 16 * j) * K + k0, K);
      acc[0][j] = wmma_f16(f0, b, acc[0][j]);
      acc[1][j] = wmma_f16(f1, b, acc[1][j]);
    }
  }

  unsigned lane = threadIdx.x & 31u;
  int cn = lane & 15;
  int rm = (lane >> 4) * 8;
#pragma unroll
  for (int i = 0; i < 2; ++i)
#pragma unroll
    for (int j = 0; j < 4; ++j) {
      int mm = m0 + i * 16 + rm;
      int nn = n0 + j * 16 + cn;
      float bz = bias[nn];
#pragma unroll
      for (int r = 0; r < 8; ++r) {
        size_t idx = (size_t)(mm + r) * N + nn;
        Out[idx] = acc[i][j][r] + bz + resid[idx];
      }
    }
}

// ---------------------------------------------------------------------------
// In-place LayerNorm over rows of 1024
// ---------------------------------------------------------------------------
__global__ __launch_bounds__(256) void ln_kernel(float* __restrict__ out,
                                                 const float* __restrict__ gamma,
                                                 const float* __restrict__ beta) {
  float* p = out + (size_t)blockIdx.x * 1024;
  int t = threadIdx.x;
  float v[4], s = 0.f, ss = 0.f;
#pragma unroll
  for (int i = 0; i < 4; ++i) {
    v[i] = p[t + i * 256];
    s += v[i];
    ss += v[i] * v[i];
  }
#pragma unroll
  for (int off = 1; off < 32; off <<= 1) {
    s += __shfl_xor(s, off, 32);
    ss += __shfl_xor(ss, off, 32);
  }
  __shared__ float red[16];
  if ((t & 31) == 0) { red[t >> 5] = s; red[8 + (t >> 5)] = ss; }
  __syncthreads();
  float S = 0.f, SS = 0.f;
#pragma unroll
  for (int w = 0; w < 8; ++w) { S += red[w]; SS += red[8 + w]; }
  float mu = S * (1.f / 1024.f);
  float inv = rsqrtf(SS * (1.f / 1024.f) - mu * mu + 1e-5f);
#pragma unroll
  for (int i = 0; i < 4; ++i) {
    int c = t + i * 256;
    p[c] = (v[i] - mu) * inv * gamma[c] + beta[c];
  }
}

// ---------------------------------------------------------------------------
extern "C" void kernel_launch(void* const* d_in, const int* in_sizes, int n_in,
                              void* d_out, int out_size, void* d_ws, size_t ws_size,
                              hipStream_t stream) {
  (void)in_sizes; (void)n_in; (void)out_size; (void)ws_size;
  const float* val           = (const float*)d_in[0];
  const unsigned char* kpm   = (const unsigned char*)d_in[1];
  const float* w_qkv         = (const float*)d_in[2];
  const float* b_qkv         = (const float*)d_in[3];
  const float* w_out         = (const float*)d_in[4];
  const float* b_out         = (const float*)d_in[5];
  const float* ln_gamma      = (const float*)d_in[6];
  const float* ln_beta       = (const float*)d_in[7];
  float* out                 = (float*)d_out;

  const size_t NX  = (size_t)16384 * 1024;      // val elements
  const size_t NWQ = (size_t)3072 * 1024;
  const size_t NWO = (size_t)1024 * 1024;
  const size_t NQK = (size_t)16384 * 3072;

  _Float16* ws     = (_Float16*)d_ws;
  _Float16* x_h    = ws;
  _Float16* wqkv_h = x_h + NX;
  _Float16* wout_h = wqkv_h + NWQ;
  _Float16* qkv_h  = wout_h + NWO;
  _Float16* attn_h = qkv_h + NQK;

  cvt_kernel<<<(int)(NX / 256), 256, 0, stream>>>(val, x_h, (int)NX);
  cvt_kernel<<<(int)(NWQ / 256), 256, 0, stream>>>(w_qkv, wqkv_h, (int)NWQ);
  cvt_kernel<<<(int)(NWO / 256), 256, 0, stream>>>(w_out, wout_h, (int)NWO);

  qkv_gemm_kernel<<<dim3(128, 24), 256, 0, stream>>>(x_h, wqkv_h, b_qkv, qkv_h);
  attn_kernel<<<1024, 256, 0, stream>>>(qkv_h, kpm, attn_h);
  proj_gemm_kernel<<<dim3(128, 8), 256, 0, stream>>>(attn_h, wout_h, b_out, val, out);
  ln_kernel<<<16384, 256, 0, stream>>>(out, ln_gamma, ln_beta);
}
